// SympleAgent_65317862637595
// MI455X (gfx1250) — matrix operand order repaired and checked by reading, not compiled, not run on hardware
//
#include <hip/hip_runtime.h>
#include <hip/hip_bf16.h>

typedef __bf16 bf16_t;
typedef __attribute__((ext_vector_type(16))) __bf16 v16bf;
typedef __attribute__((ext_vector_type(8)))  __bf16 v8bf;
typedef __attribute__((ext_vector_type(8)))  float  v8f;

#define BB 64
#define HH 512
#define VV 512
#define GG 512
#define NNODE 1023
#define DD 9
#define OUTW 1096   // 64 + 4 + 5 + 1023

__device__ __forceinline__ float sigm(float x) { return 1.0f / (1.0f + __expf(-x)); }

__device__ __forceinline__ v16bf frag16(const bf16_t* p0, const bf16_t* p1) {
    v8bf lo = *(const v8bf*)p0;
    v8bf hi = *(const v8bf*)p1;
    return __builtin_shufflevector(lo, hi, 0, 1, 2, 3, 4, 5, 6, 7,
                                   8, 9, 10, 11, 12, 13, 14, 15);
}

__device__ __forceinline__ v8f wmma_bf16(v16bf a, v16bf b, v8f c) {
    return __builtin_amdgcn_wmma_f32_16x16x32_bf16(false, a, false, b,
                                                   (short)0, c, false, false);
}

// ---------------------------------------------------------------------------
// Generic dual-input WMMA GEMM:  C = A0@B0t + A1@B1t (+bias) (+relu).
// A: bf16 row-major (M x K); B*t: bf16 TRANSPOSED (N x K, row stride ldb).
// Block tile 64(M) x 128(N); 8 waves, each wave computes a 2x2 grid of 16x16
// sub-tiles (4 wmma / K-step).  LDS double-buffered; next tile is prefetched
// into registers while the current tile's WMMAs execute (1 barrier / K-step).
// Requirements: M%64==0, N%128==0, K%32==0 (all call sites comply).
// ---------------------------------------------------------------------------
__global__ __launch_bounds__(256) void wmma_gemm(
    const bf16_t* __restrict__ A0, long lda0,
    const bf16_t* __restrict__ B0t, long ldb0, int K0,
    const bf16_t* __restrict__ A1, long lda1,
    const bf16_t* __restrict__ B1t, long ldb1, int K1,
    const float* __restrict__ bias,
    float* __restrict__ C, long ldc, bf16_t* __restrict__ Cbf,
    int M, int Ncol, int relu)
{
    __shared__ __align__(16) bf16_t sA[2][64 * 32];    // [row][k]
    __shared__ __align__(16) bf16_t sB[2][128 * 32];   // [col][k] (transposed tile)

    const int bm   = blockIdx.y * 64;
    const int bn   = blockIdx.x * 128;
    const int tid  = threadIdx.x;
    const int wave = tid >> 5;
    const int lane = tid & 31;
    const int mw   = wave & 1;          // m sub-tiles: mw*2 + {0,1}
    const int nw   = wave >> 1;         // n sub-tiles: nw*2 + {0,1}
    const int half = lane >> 4;
    const int l16  = lane & 15;

    // staging: A tile 64x32 (8 bf16/thread), B tile 128x32 (16 bf16/thread)
    const int arow = tid >> 2, acol = (tid & 3) * 8;
    const int brow = tid >> 1, bcol = (tid & 1) * 16;

    const int nk0 = K0 >> 5;
    const int nk1 = (A1 != nullptr) ? (K1 >> 5) : 0;
    const int nk  = nk0 + nk1;

    auto loadA = [&](int s, v8bf& ra) {
        if (s < nk0)
            ra = *(const v8bf*)(A0 + (long)(bm + arow) * lda0 + (s << 5) + acol);
        else
            ra = *(const v8bf*)(A1 + (long)(bm + arow) * lda1 + ((s - nk0) << 5) + acol);
    };
    auto loadB = [&](int s, v8bf& rb0, v8bf& rb1) {
        const bf16_t* p = (s < nk0)
            ? B0t + (long)(bn + brow) * ldb0 + (s << 5) + bcol
            : B1t + (long)(bn + brow) * ldb1 + ((s - nk0) << 5) + bcol;
        rb0 = *(const v8bf*)p;
        rb1 = *(const v8bf*)(p + 8);
    };

    v8f acc[2][2] = {};
    v8bf ra, rb0, rb1;

    loadA(0, ra);
    loadB(0, rb0, rb1);
    *(v8bf*)&sA[0][arow * 32 + acol]     = ra;
    *(v8bf*)&sB[0][brow * 32 + bcol]     = rb0;
    *(v8bf*)&sB[0][brow * 32 + bcol + 8] = rb1;
    __syncthreads();

    for (int s = 0; s < nk; ++s) {
        const int  cur  = s & 1;
        const bool more = (s + 1) < nk;
        if (more) { loadA(s + 1, ra); loadB(s + 1, rb0, rb1); }   // overlaps WMMA

        // A fragments (rows mw*32 + {0,16} + l16): two b128 LDS loads each
        const bf16_t* a0p = &sA[cur][(mw * 32 + l16) * 32];
        const bf16_t* a1p = a0p + 16 * 32;
        v16bf af0 = frag16(a0p + half * 8, a0p + 16 + half * 8);
        v16bf af1 = frag16(a1p + half * 8, a1p + 16 + half * 8);

        // B fragments (cols nw*32 + {0,16} + l16): two b128 LDS loads each
        const bf16_t* b0p = &sB[cur][(nw * 32 + l16) * 32 + half * 8];
        const bf16_t* b1p = b0p + 16 * 32;
        v16bf bf0 = frag16(b0p, b0p + 16);
        v16bf bf1 = frag16(b1p, b1p + 16);

        acc[0][0] = wmma_bf16(af0, bf0, acc[0][0]);
        acc[0][1] = wmma_bf16(af0, bf1, acc[0][1]);
        acc[1][0] = wmma_bf16(af1, bf0, acc[1][0]);
        acc[1][1] = wmma_bf16(af1, bf1, acc[1][1]);

        if (more) {
            const int nxt = cur ^ 1;
            *(v8bf*)&sA[nxt][arow * 32 + acol]     = ra;
            *(v8bf*)&sB[nxt][brow * 32 + bcol]     = rb0;
            *(v8bf*)&sB[nxt][brow * 32 + bcol + 8] = rb1;
            __syncthreads();
        }
    }

    // C/D layout: VGPR i -> row i + 8*half, col = l16 within each 16x16 tile
#pragma unroll
    for (int mi = 0; mi < 2; ++mi) {
#pragma unroll
        for (int ni = 0; ni < 2; ++ni) {
            const int rbase = bm + (mw * 2 + mi) * 16 + 8 * half;
            const int c     = bn + (nw * 2 + ni) * 16 + l16;
            const float bv  = bias ? bias[c] : 0.0f;
#pragma unroll
            for (int i = 0; i < 8; ++i) {
                const int r = rbase + i;
                float v = acc[mi][ni][i] + bv;
                if (relu) v = fmaxf(v, 0.f);
                C[(long)r * ldc + c] = v;
                if (Cbf) Cbf[(long)r * ldc + c] = (bf16_t)v;
            }
        }
    }
}

// --------------------------- small helper kernels ---------------------------

__global__ void k_cvt(const float* __restrict__ in, bf16_t* __restrict__ out, long n) {
    long i = (long)blockIdx.x * blockDim.x + threadIdx.x;
    if (i < n) out[i] = (bf16_t)in[i];
}

// Transposing convert: in (K, N) f32 row-major -> out (N, K) bf16 row-major
__global__ void k_cvt_T(const float* __restrict__ in, bf16_t* __restrict__ out,
                        int K, int N) {
    long i = (long)blockIdx.x * blockDim.x + threadIdx.x;
    if (i >= (long)K * N) return;
    int n  = (int)(i / K);
    int kk = (int)(i % K);
    out[i] = (bf16_t)in[(long)kk * N + n];
}

__global__ void k_zero_f32(float* p, long n) {
    long i = (long)blockIdx.x * blockDim.x + threadIdx.x;
    if (i < n) p[i] = 0.0f;
}

// Transposed block-diagonal [[U_f[0],0],[0,U_f[1]]]^T as (1024,1024) bf16
__global__ void k_uf_blockdiag_T(const float* __restrict__ Uf, bf16_t* __restrict__ out) {
    long i = (long)blockIdx.x * blockDim.x + threadIdx.x;
    if (i >= (long)1024 * 1024) return;
    int n = (int)(i >> 10), kk = (int)(i & 1023);   // out[n][kk] = orig[kk][n]
    float v = 0.0f;
    if (kk < 512 && n < 512)        v = Uf[(long)kk * 512 + n];
    else if (kk >= 512 && n >= 512) v = Uf[(long)512 * 512 + (long)(kk - 512) * 512 + (n - 512)];
    out[i] = (bf16_t)v;
}

// Gather x[:, node0:node0+n, :] -> contiguous bf16 (B*n, V)
__global__ void k_gather_x(const float* __restrict__ x, bf16_t* __restrict__ out,
                           int n, int node0, long total) {
    long i = (long)blockIdx.x * blockDim.x + threadIdx.x;
    if (i >= total) return;
    long r = i / VV; int v = (int)(i % VV);
    int b = (int)(r / n), node = (int)(r % n);
    out[i] = (bf16_t)x[((long)b * NNODE + node0 + node) * VV + v];
}

// TreeLSTM elementwise: gates, cell update, h out (bf16 level buffer + hs_bf)
__global__ void k_tree_elem(const float* __restrict__ iou, const float* __restrict__ fx,
                            const float* __restrict__ fh, const float* __restrict__ c_child,
                            float* __restrict__ c_out, bf16_t* __restrict__ hs_bf,
                            bf16_t* __restrict__ h_bf, int n, int node0, long total) {
    long i = (long)blockIdx.x * blockDim.x + threadIdx.x;
    if (i >= total) return;
    long r = i >> 9; int hc = (int)(i & 511);
    const float* ir = iou + r * 1536;
    float si = sigm(ir[hc]);
    float so = sigm(ir[512 + hc]);
    float tu = tanhf(ir[1024 + hc]);
    float c = si * tu;
    if (c_child) {
        float fxv = fx[r * 512 + hc];
        float f0 = sigm(fxv + fh[r * 1024 + hc]);
        float f1 = sigm(fxv + fh[r * 1024 + 512 + hc]);
        c += f0 * c_child[r * 1024 + hc] + f1 * c_child[r * 1024 + 512 + hc];
    }
    float h = so * tanhf(c);
    c_out[i] = c;
    int b = (int)(r / n), node = (int)(r % n);
    hs_bf[((long)b * NNODE + node0 + node) * HH + hc] = (bf16_t)h;
    h_bf[i] = (bf16_t)h;
}

// LSTM gate update; also mirrors h to bf16 for the next step's GEMM
__global__ void k_lstm_elem(const float* __restrict__ z, float* __restrict__ h,
                            float* __restrict__ c, bf16_t* __restrict__ h_bf) {
    int i = blockIdx.x * blockDim.x + threadIdx.x;
    if (i >= BB * GG) return;
    int b = i >> 9, j = i & 511;
    const float* zr = z + (long)b * 2048;
    float ig = sigm(zr[j]);
    float fg = sigm(zr[512 + j]);
    float gg = tanhf(zr[1024 + j]);
    float og = sigm(zr[1536 + j]);
    float cv = fg * c[i] + ig * gg;
    float hv = og * tanhf(cv);
    c[i] = cv; h[i] = hv; h_bf[i] = (bf16_t)hv;
}

// feat = concat(root, h_glob) as bf16; also mirrors h_glob to bf16
__global__ void k_feat(const float* __restrict__ root, const float* __restrict__ hglob,
                       bf16_t* __restrict__ feat_bf, bf16_t* __restrict__ hglob_bf) {
    int i = blockIdx.x * blockDim.x + threadIdx.x;
    if (i >= BB * 1024) return;
    int b = i >> 10, j = i & 1023;
    float v = (j < 512) ? root[b * 512 + j] : hglob[b * 512 + (j - 512)];
    feat_bf[i] = (bf16_t)v;
    if (j >= 512) hglob_bf[b * 512 + (j - 512)] = (bf16_t)v;
}

// Tiny-N head: out[b, col0+j] = A[b,:]@W[:,j] + bias[j]   (f32 inputs)
__global__ void k_small_head(const float* __restrict__ A, const float* __restrict__ W,
                             const float* __restrict__ bias, float* __restrict__ out,
                             int nOut, int col0) {
    int i = blockIdx.x * blockDim.x + threadIdx.x;
    if (i >= BB * nOut) return;
    int b = i / nOut, j = i % nOut;
    float s = bias[j];
    const float* ar = A + (long)b * 512;
    for (int k = 0; k < 512; ++k) s += ar[k] * W[(long)k * nOut + j];
    out[(long)b * OUTW + col0 + j] = s;
}

// tp1 = relu(tp1 + t0[batch] + b1)
__global__ void k_tp_bias_relu(float* __restrict__ tp1, const float* __restrict__ t0,
                               const float* __restrict__ b1, long total) {
    long i = (long)blockIdx.x * blockDim.x + threadIdx.x;
    if (i >= total) return;
    long r = i >> 9; int j = (int)(i & 511);
    int b = (int)(r / NNODE);
    float v = tp1[i] + t0[(long)b * 512 + j] + b1[j];
    tp1[i] = fmaxf(v, 0.0f);
}

// tp_scores[b, t] -> out col 73+t
__global__ void k_tp_dot(const float* __restrict__ tp1, const float* __restrict__ W2,
                         const float* __restrict__ b2, float* __restrict__ out) {
    int r = blockIdx.x * blockDim.x + threadIdx.x;
    if (r >= BB * NNODE) return;
    float s = b2[0];
    const float* ar = tp1 + (long)r * 512;
    for (int k = 0; k < 512; ++k) s += ar[k] * W2[k];
    int b = r / NNODE, t = r % NNODE;
    out[(long)b * OUTW + 73 + t] = s;
}

// ------------------------------- host side ----------------------------------

static inline void launch_gemm(hipStream_t s,
    const bf16_t* A0, long lda0, const bf16_t* B0t, long ldb0, int K0,
    const bf16_t* A1, long lda1, const bf16_t* B1t, long ldb1, int K1,
    const float* bias, float* C, long ldc, bf16_t* Cbf, int M, int N, int relu)
{
    dim3 grid((unsigned)(N / 128), (unsigned)(M / 64));
    wmma_gemm<<<grid, 256, 0, s>>>(A0, lda0, B0t, ldb0, K0, A1, lda1, B1t, ldb1, K1,
                                   bias, C, ldc, Cbf, M, N, relu);
}

static inline void launch_cvtT(hipStream_t s, const float* in, bf16_t* out, int K, int N) {
    long n = (long)K * N;
    k_cvt_T<<<(unsigned)((n + 255) / 256), 256, 0, s>>>(in, out, K, N);
}

static inline void launch_cvt(hipStream_t s, const float* in, bf16_t* out, long n) {
    k_cvt<<<(unsigned)((n + 255) / 256), 256, 0, s>>>(in, out, n);
}

extern "C" void kernel_launch(void* const* d_in, const int* in_sizes, int n_in,
                              void* d_out, int out_size, void* d_ws, size_t ws_size,
                              hipStream_t stream) {
    (void)in_sizes; (void)n_in; (void)out_size; (void)ws_size;
    const float* x      = (const float*)d_in[0];
    const float* W_iou  = (const float*)d_in[1];
    const float* U_iou  = (const float*)d_in[2];
    const float* b_iou  = (const float*)d_in[3];
    const float* W_f    = (const float*)d_in[4];
    const float* U_f    = (const float*)d_in[5];
    const float* b_f    = (const float*)d_in[6];
    const float* W_ih   = (const float*)d_in[7];
    const float* W_hh   = (const float*)d_in[8];
    const float* b_lstm = (const float*)d_in[9];
    const float* ffn_W1 = (const float*)d_in[10];
    const float* ffn_b1 = (const float*)d_in[11];
    const float* ffn_W2 = (const float*)d_in[12];
    const float* ffn_b2 = (const float*)d_in[13];
    const float* act_W1 = (const float*)d_in[14];
    const float* act_b1 = (const float*)d_in[15];
    const float* act_W2 = (const float*)d_in[16];
    const float* act_b2 = (const float*)d_in[17];
    const float* hl_W1  = (const float*)d_in[18];
    const float* hl_b1  = (const float*)d_in[19];
    const float* hl_W2  = (const float*)d_in[20];
    const float* hl_b2  = (const float*)d_in[21];
    const float* int_W1 = (const float*)d_in[22];
    const float* int_b1 = (const float*)d_in[23];
    const float* int_W2 = (const float*)d_in[24];
    const float* int_b2 = (const float*)d_in[25];
    const float* tp_W1  = (const float*)d_in[26];
    const float* tp_b1  = (const float*)d_in[27];
    const float* tp_W2  = (const float*)d_in[28];
    const float* tp_b2  = (const float*)d_in[29];
    float* out = (float*)d_out;

    // --- workspace arena ---
    char* ws = (char*)d_ws;
    size_t off = 0;
    auto alloc = [&](size_t bytes) -> void* {
        void* p = ws + off;
        off = (off + bytes + 255) & ~(size_t)255;
        return p;
    };
    auto abf = [&](long n) -> bf16_t* { return (bf16_t*)alloc((size_t)n * 2); };
    auto af32 = [&](long n) -> float* { return (float*)alloc((size_t)n * 4); };

    // all weight matrices stored TRANSPOSED (N x K) in bf16
    bf16_t* W_iou_t = abf(1536L * 512);
    bf16_t* U_iou_t = abf(1536L * 1024);
    bf16_t* W_f_t   = abf(512L * 512);
    bf16_t* UFbd_t  = abf(1024L * 1024);
    bf16_t* W_ih_t  = abf(2048L * 512);
    bf16_t* W_hh_t  = abf(2048L * 512);
    bf16_t* ffnW1_t = abf(512L * 512);
    bf16_t* ffnW2_t = abf(512L * 512);
    bf16_t* actW1_t = abf(512L * 1024);
    bf16_t* hlW1_t  = abf(512L * 1024);
    bf16_t* intW1_t = abf(512L * 1024);
    bf16_t* tpW1_t  = abf(512L * 1024);

    bf16_t* xbf   = abf((long)BB * 512 * VV);
    bf16_t* hbfA  = abf((long)BB * 512 * HH);
    bf16_t* hbfB  = abf((long)BB * 512 * HH);
    float*  cA    = af32((long)BB * 512 * HH);
    float*  cB    = af32((long)BB * 512 * HH);
    bf16_t* hsbf  = abf((long)BB * NNODE * HH);
    float*  iou   = af32((long)BB * 512 * 1536);   // also reused as tp1 (smaller)
    float*  fx    = af32((long)BB * 512 * HH);
    float*  fh    = af32((long)BB * 512 * 1024);
    float*  z     = af32((long)BB * 2048);
    float*  hsc   = af32((long)BB * GG);
    float*  csc   = af32((long)BB * GG);
    bf16_t* hscb  = abf((long)BB * GG);
    float*  r1    = af32((long)BB * HH);
    bf16_t* r1b   = abf((long)BB * HH);
    float*  root  = af32((long)BB * HH);
    bf16_t* featb = abf((long)BB * 1024);
    bf16_t* hglb  = abf((long)BB * GG);
    float*  g1    = af32((long)BB * HH);
    float*  h1    = af32((long)BB * HH);
    float*  i1    = af32((long)BB * HH);
    float*  t0    = af32((long)BB * HH);

    // --- weight conversion (f32 -> bf16, transposed) ---
    launch_cvtT(stream, W_iou, W_iou_t, 512, 1536);
    launch_cvtT(stream, U_iou, U_iou_t, 1024, 1536);
    launch_cvtT(stream, W_f,   W_f_t,   512, 512);
    launch_cvtT(stream, W_ih,  W_ih_t,  512, 2048);
    launch_cvtT(stream, W_hh,  W_hh_t,  512, 2048);
    launch_cvtT(stream, ffn_W1, ffnW1_t, 512, 512);
    launch_cvtT(stream, ffn_W2, ffnW2_t, 512, 512);
    launch_cvtT(stream, act_W1, actW1_t, 1024, 512);
    launch_cvtT(stream, hl_W1,  hlW1_t,  1024, 512);
    launch_cvtT(stream, int_W1, intW1_t, 1024, 512);
    launch_cvtT(stream, tp_W1,  tpW1_t,  1024, 512);
    k_uf_blockdiag_T<<<(1024u * 1024u) / 256u, 256, 0, stream>>>(U_f, UFbd_t);

    // --- TreeLSTM, bottom-up ---
    const bf16_t* hchild = nullptr;
    const float*  cchild = nullptr;
    for (int l = DD; l >= 0; --l) {
        const int n = 1 << l;
        const int node0 = n - 1;
        const int M = BB * n;                 // multiple of 64
        const long tot_x = (long)M * VV;
        k_gather_x<<<(unsigned)((tot_x + 255) / 256), 256, 0, stream>>>(x, xbf, n, node0, tot_x);

        // iou = x_l@W_iou (+ hcat@U_iou) + b_iou   (dual-K in one launch)
        launch_gemm(stream, xbf, VV, W_iou_t, 512, VV,
                    hchild, 1024, U_iou_t, 1024, hchild ? 1024 : 0,
                    b_iou, iou, 1536, nullptr, M, 1536, 0);
        if (hchild) {
            launch_gemm(stream, xbf, VV, W_f_t, 512, VV,
                        nullptr, 0, nullptr, 0, 0,
                        b_f, fx, 512, nullptr, M, 512, 0);
            launch_gemm(stream, hchild, 1024, UFbd_t, 1024, 1024,
                        nullptr, 0, nullptr, 0, 0,
                        nullptr, fh, 1024, nullptr, M, 1024, 0);
        }
        bf16_t* hout = ((DD - l) & 1) ? hbfB : hbfA;
        float*  cout = ((DD - l) & 1) ? cB : cA;
        const long tot_h = (long)M * HH;
        k_tree_elem<<<(unsigned)((tot_h + 255) / 256), 256, 0, stream>>>(
            iou, fx, fh, cchild, cout, hsbf, hout, n, node0, tot_h);
        hchild = hout; cchild = cout;
    }

    // --- global LSTM scan over the 1023 node states ---
    k_zero_f32<<<(BB * GG) / 256, 256, 0, stream>>>(hsc, (long)BB * GG);
    k_zero_f32<<<(BB * GG) / 256, 256, 0, stream>>>(csc, (long)BB * GG);
    launch_cvt(stream, hsc, hscb, (long)BB * GG);   // bf16 h starts at zero
    for (int t = 0; t < NNODE; ++t) {
        // z = x_t@W_ih + h@W_hh + b_lstm  (x_t = hs[:, t] via strided A)
        launch_gemm(stream, hsbf + (long)t * HH, (long)NNODE * HH, W_ih_t, 512, HH,
                    hscb, GG, W_hh_t, 512, GG,
                    b_lstm, z, 2048, nullptr, BB, 2048, 0);
        k_lstm_elem<<<(BB * GG) / 256, 256, 0, stream>>>(z, hsc, csc, hscb);
    }

    // --- root FFN: hs[:,0] (strided) -> relu -> f32+bf16 -> second GEMM ---
    launch_gemm(stream, hsbf, (long)NNODE * HH, ffnW1_t, 512, HH,
                nullptr, 0, nullptr, 0, 0, ffn_b1, r1, 512, r1b, BB, 512, 1);
    launch_gemm(stream, r1b, 512, ffnW2_t, 512, 512,
                nullptr, 0, nullptr, 0, 0, ffn_b2, root, 512, nullptr, BB, 512, 0);

    // feat = concat(root, h_glob)
    k_feat<<<(BB * 1024) / 256, 256, 0, stream>>>(root, hsc, featb, hglb);

    // act head -> out cols [0,64):  g1 = relu(feat@act_W1+b1); then scalar N=64
    launch_gemm(stream, featb, 1024, actW1_t, 1024, 1024,
                nullptr, 0, nullptr, 0, 0, act_b1, g1, 512, nullptr, BB, 512, 1);
    k_small_head<<<16, 256, 0, stream>>>(g1, act_W2, act_b2, out, 64, 0);

    // hl head -> out cols [64,68)
    launch_gemm(stream, featb, 1024, hlW1_t, 1024, 1024,
                nullptr, 0, nullptr, 0, 0, hl_b1, h1, 512, nullptr, BB, 512, 1);
    k_small_head<<<1, 256, 0, stream>>>(h1, hl_W2, hl_b2, out, 4, 64);

    // int head -> out cols [68,73)
    launch_gemm(stream, featb, 1024, intW1_t, 1024, 1024,
                nullptr, 0, nullptr, 0, 0, int_b1, i1, 512, nullptr, BB, 512, 1);
    k_small_head<<<2, 256, 0, stream>>>(i1, int_W2, int_b2, out, 5, 68);

    // tp head: t0 = h_glob @ tp_W1[:G]; tp1 = hs_flat @ tp_W1[G:]
    // tpW1_t is (512 out, 1024 k): k 0..511 = h_glob part, k 512..1023 = hs part
    launch_gemm(stream, hglb, GG, tpW1_t, 1024, GG,
                nullptr, 0, nullptr, 0, 0, nullptr, t0, 512, nullptr, BB, 512, 0);
    float* tp1 = iou;   // reuse big scratch (tree done)
    launch_gemm(stream, hsbf, HH, tpW1_t + 512, 1024, HH,
                nullptr, 0, nullptr, 0, 0, nullptr, tp1, 512, nullptr, BB * NNODE, 512, 0);
    const long tot_tp = (long)BB * NNODE * 512;
    k_tp_bias_relu<<<(unsigned)((tot_tp + 255) / 256), 256, 0, stream>>>(tp1, t0, tp_b1, tot_tp);
    k_tp_dot<<<(unsigned)((BB * NNODE + 255) / 256), 256, 0, stream>>>(tp1, tp_W2, tp_b2, out);
}
